// MultiHead_Attention_80522046865890
// MI455X (gfx1250) — matrix operand order repaired and checked by reading, not compiled, not run on hardware
//
#include <hip/hip_runtime.h>
#include <hip/hip_bf16.h>

typedef __attribute__((ext_vector_type(16))) _Float16 v16h;
typedef __attribute__((ext_vector_type(8)))  _Float16 v8h;
typedef __attribute__((ext_vector_type(4)))  _Float16 v4h;
typedef __attribute__((ext_vector_type(8)))  float    v8f;
typedef __attribute__((ext_vector_type(4)))  float    f32x4;

#define B_SZ 4
#define L_SZ 2048
#define D_SZ 512
#define H_SZ 8
#define E_SZ 64
#define SCALE 0.125f   // 1/sqrt(64)

union V16U { v16h v; v8h h[2]; };

// ---------------- WMMA fragment helpers (wave32, ISA 7.12.2 layouts) --------
// A fragment 16x32 f16: lane l holds row M=l%16; halves are two contiguous
// 8-half runs at K = kb..kb+7 and kb+16..kb+23, kb = k0 + 8*(lane>=16).
// With LDS stride a multiple of 8 halves this is exactly 2x ds_load_b128.
__device__ __forceinline__ v16h frag_a_lds(const _Float16* lds, int ld,
                                           int row0, int k0, int lane) {
    const int m  = row0 + (lane & 15);
    const int kb = k0 + ((lane & 16) ? 8 : 0);
    const _Float16* p = lds + m * ld + kb;
    V16U u;
    u.h[0] = *(const v8h*)(p);
    u.h[1] = *(const v8h*)(p + 16);
    return u.v;
}

// B fragment 32x16 f16 from an N-major tile (src[n][k], row-major, stride ld):
// lane holds column N=lane%16, two contiguous 8-half runs K = kb..kb+7 and
// kb+16..kb+23, kb = k0 + 8*(lane>=16)  ->  2x aligned ds_load_b128.
__device__ __forceinline__ v16h frag_bt_lds(const _Float16* lds, int ld,
                                            int k0, int n0, int lane) {
    const int n  = n0 + (lane & 15);
    const int kb = k0 + ((lane & 16) ? 8 : 0);
    const _Float16* p = lds + n * ld + kb;
    V16U u;
    u.h[0] = *(const v8h*)(p);
    u.h[1] = *(const v8h*)(p + 16);
    return u.v;
}

__device__ __forceinline__ v8f wmma16x16x32(v16h a, v16h b, v8f c) {
    return __builtin_amdgcn_wmma_f32_16x16x32_f16(false, a, false, b,
                                                  (short)0, c, false, false);
}

// C/D layout: VGPR r: lanes 0-15 -> (M=r, N=lane); lanes 16-31 -> (M=r+8).
__device__ __forceinline__ void store_f16_bias(_Float16* O, const float* bias,
                                               int grow0, int gcol0, int lane,
                                               v8f c) {
    const int n  = gcol0 + (lane & 15);
    const int mb = grow0 + ((lane & 16) ? 8 : 0);
    const float bb = bias[n];
#pragma unroll
    for (int r = 0; r < 8; ++r)
        O[(size_t)(mb + r) * D_SZ + n] = (_Float16)(c[r] + bb);
}

// V stored transposed: Vt[(b*H+h)*E + e][l].  The 8 accumulator rows are
// contiguous l values -> one 16-byte global store per lane-tile.
__device__ __forceinline__ void store_vt_bias(_Float16* Vt, const float* bias,
                                              int grow0, int gcol0, int lane,
                                              v8f c) {
    const int n  = gcol0 + (lane & 15);          // n = h*E + e
    const int mb = grow0 + ((lane & 16) ? 8 : 0); // mb = b*L + l (8-aligned)
    const int b  = mb / L_SZ, l = mb % L_SZ;
    const int h  = n / E_SZ,  e = n % E_SZ;
    const float bb = bias[n];
    _Float16* p = Vt + (((size_t)(b * H_SZ + h) * E_SZ + e) * L_SZ + l);
    v8h hh;
#pragma unroll
    for (int r = 0; r < 8; ++r) hh[r] = (_Float16)(c[r] + bb);
    *(v8h*)p = hh;
}

// ---------------- Kernel 0: one-time weight transpose W -> f16 Wt[n][k] -----
// Tiny (3 x 512 x 512); hoists the transpose out of the projection hot loop.
__global__ __launch_bounds__(256)
void wt_kernel(const float* __restrict__ Wq, const float* __restrict__ Wk,
               const float* __restrict__ Wv, _Float16* __restrict__ wt)
{
    __shared__ __align__(16) _Float16 t[64][72];   // [k][n] staging
    const float* W = blockIdx.z == 0 ? Wq : (blockIdx.z == 1 ? Wk : Wv);
    _Float16* Wt = wt + (size_t)blockIdx.z * D_SZ * D_SZ;
    const int k0 = blockIdx.x * 64, n0 = blockIdx.y * 64;
    const int tid = threadIdx.x;

#pragma unroll
    for (int i = 0; i < 4; ++i) {          // 64x64 f32 tile: 1024 f32x4 chunks
        const int idx = tid + i * 256;
        const int r = idx >> 4, c4 = (idx & 15) << 2;   // r = k, c4 = n
        f32x4 x = *(const f32x4*)&W[(size_t)(k0 + r) * D_SZ + n0 + c4];
        v4h hh;
        hh[0] = (_Float16)x[0]; hh[1] = (_Float16)x[1];
        hh[2] = (_Float16)x[2]; hh[3] = (_Float16)x[3];
        *(v4h*)&t[r][c4] = hh;
    }
    __syncthreads();

#pragma unroll
    for (int i = 0; i < 2; ++i) {          // write Wt rows: 512 v8h chunks
        const int idx = tid + i * 256;
        const int n = idx >> 3, c8 = (idx & 7) << 3;    // n row, k run of 8
        v8h hh;
#pragma unroll
        for (int q = 0; q < 8; ++q) hh[q] = t[c8 + q][n];
        *(v8h*)&Wt[(size_t)(n0 + n) * D_SZ + k0 + c8] = hh;
    }
}

// ---------------- Kernel 1: Q/K/V projection (X @ W + b) -> f16 ws ----------
__global__ __launch_bounds__(256)
void qkv_proj_kernel(const float* __restrict__ q_in, const float* __restrict__ k_in,
                     const float* __restrict__ v_in,
                     const _Float16* __restrict__ wt,
                     const float* __restrict__ bq, const float* __restrict__ bk,
                     const float* __restrict__ bv,
                     _Float16* __restrict__ qo, _Float16* __restrict__ ko,
                     _Float16* __restrict__ vt)
{
    __shared__ __align__(16) _Float16 Ash[128][40];  // [m][k], stride 8-mult
    __shared__ __align__(16) _Float16 Bsh[64][40];   // [n][k] (N-major)

    const int which = blockIdx.z;
    const float*    X    = which == 0 ? q_in : (which == 1 ? k_in : v_in);
    const _Float16* Wt   = wt + (size_t)which * D_SZ * D_SZ;
    const float*    bias = which == 0 ? bq : (which == 1 ? bk : bv);

    const int tid  = threadIdx.x;
    const int lane = tid & 31;
    const int wave = tid >> 5;
    const int row_blk = blockIdx.x * 128;
    const int col_blk = blockIdx.y * 64;
    const int wrow = (wave >> 1) * 32;   // 0..96
    const int wcol = (wave & 1) * 32;    // 0,32

    v8f c00 = {}, c01 = {}, c10 = {}, c11 = {};

    for (int k0 = 0; k0 < D_SZ; k0 += 32) {
#pragma unroll
        for (int i = 0; i < 4; ++i) {            // A: 128x32 f32 -> 1024 f32x4
            const int idx = tid + i * 256;
            const int r = idx >> 3, c4 = (idx & 7) << 2;
            f32x4 x = *(const f32x4*)&X[(size_t)(row_blk + r) * D_SZ + k0 + c4];
            v4h hh;
            hh[0] = (_Float16)x[0]; hh[1] = (_Float16)x[1];
            hh[2] = (_Float16)x[2]; hh[3] = (_Float16)x[3];
            *(v4h*)&Ash[r][c4] = hh;
        }
        {                                        // Wt: 64(n) x 32(k), 256 v8h
            const int n = tid >> 2, c8 = (tid & 3) << 3;
            *(v8h*)&Bsh[n][c8] =
                *(const v8h*)&Wt[(size_t)(col_blk + n) * D_SZ + k0 + c8];
        }
        __syncthreads();

        const v16h a0 = frag_a_lds (&Ash[0][0], 40, wrow,      0, lane);
        const v16h a1 = frag_a_lds (&Ash[0][0], 40, wrow + 16, 0, lane);
        const v16h b0 = frag_bt_lds(&Bsh[0][0], 40, 0, wcol,      lane);
        const v16h b1 = frag_bt_lds(&Bsh[0][0], 40, 0, wcol + 16, lane);

        c00 = wmma16x16x32(a0, b0, c00);
        c01 = wmma16x16x32(a0, b1, c01);
        c10 = wmma16x16x32(a1, b0, c10);
        c11 = wmma16x16x32(a1, b1, c11);
        __syncthreads();
    }

    if (which < 2) {
        _Float16* O = which == 0 ? qo : ko;
        store_f16_bias(O, bias, row_blk + wrow,      col_blk + wcol,      lane, c00);
        store_f16_bias(O, bias, row_blk + wrow,      col_blk + wcol + 16, lane, c01);
        store_f16_bias(O, bias, row_blk + wrow + 16, col_blk + wcol,      lane, c10);
        store_f16_bias(O, bias, row_blk + wrow + 16, col_blk + wcol + 16, lane, c11);
    } else {
        store_vt_bias(vt, bias, row_blk + wrow,      col_blk + wcol,      lane, c00);
        store_vt_bias(vt, bias, row_blk + wrow,      col_blk + wcol + 16, lane, c01);
        store_vt_bias(vt, bias, row_blk + wrow + 16, col_blk + wcol,      lane, c10);
        store_vt_bias(vt, bias, row_blk + wrow + 16, col_blk + wcol + 16, lane, c11);
    }
}

// ---------------- Kernel 2: scores = scale * Q @ K^T ------------------------
__global__ __launch_bounds__(256)
void scores_kernel(const _Float16* __restrict__ Q, const _Float16* __restrict__ K,
                   float* __restrict__ attn)
{
    __shared__ __align__(16) _Float16 Qsh[64][72];  // [l][e]
    __shared__ __align__(16) _Float16 Ksh[64][72];  // [key][e]: N-major for B

    const int tid  = threadIdx.x;
    const int lane = tid & 31;
    const int wave = tid >> 5;
    const int n_blk = blockIdx.x * 64;      // key tile
    const int m_blk = blockIdx.y * 64;      // query tile
    const int bh = blockIdx.z;
    const int b = bh / H_SZ, h = bh % H_SZ;

#pragma unroll
    for (int i = 0; i < 2; ++i) {           // 512 16-byte chunks each
        const int idx = tid + i * 256;
        const int r = idx >> 3, c8 = (idx & 7) << 3;
        *(v8h*)&Qsh[r][c8] =
            *(const v8h*)&Q[(size_t)(b * L_SZ + m_blk + r) * D_SZ + h * E_SZ + c8];
        *(v8h*)&Ksh[r][c8] =
            *(const v8h*)&K[(size_t)(b * L_SZ + n_blk + r) * D_SZ + h * E_SZ + c8];
    }
    __syncthreads();

    const int wrow = (wave >> 1) * 16;      // 0..48
    const int wcol = (wave & 1) * 32;       // 0,32

    v8f c0 = {}, c1 = {};
#pragma unroll
    for (int e0 = 0; e0 < E_SZ; e0 += 32) {
        const v16h a  = frag_a_lds (&Qsh[0][0], 72, wrow, e0, lane);
        const v16h b0 = frag_bt_lds(&Ksh[0][0], 72, e0, wcol,      lane);
        const v16h b1 = frag_bt_lds(&Ksh[0][0], 72, e0, wcol + 16, lane);
        c0 = wmma16x16x32(a, b0, c0);
        c1 = wmma16x16x32(a, b1, c1);
    }

    float* S = attn + ((size_t)bh * L_SZ + m_blk) * L_SZ + n_blk;
    const int n  = lane & 15;
    const int mb = (lane & 16) ? 8 : 0;
#pragma unroll
    for (int r = 0; r < 8; ++r) {   // streaming: keep Q/K/V hot in L2 instead
        __builtin_nontemporal_store(SCALE * c0[r],
            &S[(size_t)(wrow + mb + r) * L_SZ + wcol + n]);
        __builtin_nontemporal_store(SCALE * c1[r],
            &S[(size_t)(wrow + mb + r) * L_SZ + wcol + 16 + n]);
    }
}

// ---------------- Kernel 3: row softmax, in place ---------------------------
__global__ __launch_bounds__(256)
void softmax_kernel(float* __restrict__ attn)
{
    __shared__ float red[256];
    float* p = attn + (size_t)blockIdx.x * L_SZ;
    const int tid = threadIdx.x;

    float v[8];
    float mx = -1e30f;
#pragma unroll
    for (int i = 0; i < 8; ++i) {
        v[i] = p[tid + i * 256];
        mx = fmaxf(mx, v[i]);
    }
    red[tid] = mx;
    __syncthreads();
    for (int s = 128; s > 0; s >>= 1) {
        if (tid < s) red[tid] = fmaxf(red[tid], red[tid + s]);
        __syncthreads();
    }
    mx = red[0];
    __syncthreads();

    float sum = 0.f;
#pragma unroll
    for (int i = 0; i < 8; ++i) {
        v[i] = __expf(v[i] - mx);
        sum += v[i];
    }
    red[tid] = sum;
    __syncthreads();
    for (int s = 128; s > 0; s >>= 1) {
        if (tid < s) red[tid] += red[tid + s];
        __syncthreads();
    }
    const float inv = 1.0f / red[0];
#pragma unroll
    for (int i = 0; i < 8; ++i)
        __builtin_nontemporal_store(v[i] * inv, &p[tid + i * 256]);
}

// ---------------- Kernel 4: out = attn @ V (Vt layout) ----------------------
__global__ __launch_bounds__(256)
void av_kernel(const float* __restrict__ attn, const _Float16* __restrict__ Vt,
               float* __restrict__ out)
{
    __shared__ __align__(16) _Float16 Ash[64][40];   // [l][m]  (K = keys m)
    __shared__ __align__(16) _Float16 Vsh[64][40];   // [e][m]: N-major for B

    const int tid  = threadIdx.x;
    const int lane = tid & 31;
    const int wave = tid >> 5;
    const int m_blk = blockIdx.x * 64;  // query rows
    const int bh = blockIdx.y;
    const int b = bh / H_SZ, h = bh % H_SZ;

    const float* Arow = attn + ((size_t)bh * L_SZ + m_blk) * L_SZ;

    const int wrow = (wave >> 1) * 16;
    const int wcol = (wave & 1) * 32;

    v8f c0 = {}, c1 = {};
    for (int k0 = 0; k0 < L_SZ; k0 += 32) {
        if (k0 + 32 < L_SZ)
            __builtin_prefetch(&Arow[(size_t)(tid >> 2) * L_SZ + k0 + 32], 0, 0);
#pragma unroll
        for (int i = 0; i < 2; ++i) {        // attn tile 64x32 f32 -> f16 (NT)
            const int idx = tid + i * 256;   // 512 f32x4 chunks
            const int r = idx >> 3, c4 = (idx & 7) << 2;
            f32x4 x = __builtin_nontemporal_load(
                (const f32x4*)&Arow[(size_t)r * L_SZ + k0 + c4]);
            v4h hh;
            hh[0] = (_Float16)x[0]; hh[1] = (_Float16)x[1];
            hh[2] = (_Float16)x[2]; hh[3] = (_Float16)x[3];
            *(v4h*)&Ash[r][c4] = hh;
        }
        {                                    // Vt tile 64(e) x 32(m), 256 chunks
            const int r = tid >> 2, c8 = (tid & 3) << 3;
            *(v8h*)&Vsh[r][c8] =
                *(const v8h*)&Vt[((size_t)bh * E_SZ + r) * L_SZ + k0 + c8];
        }
        __syncthreads();

        const v16h a  = frag_a_lds (&Ash[0][0], 40, wrow, 0, lane);
        const v16h b0 = frag_bt_lds(&Vsh[0][0], 40, 0, wcol,      lane);
        const v16h b1 = frag_bt_lds(&Vsh[0][0], 40, 0, wcol + 16, lane);
        c0 = wmma16x16x32(a, b0, c0);
        c1 = wmma16x16x32(a, b1, c1);
        __syncthreads();
    }

    const int n  = lane & 15;
    const int mb = (lane & 16) ? 8 : 0;
#pragma unroll
    for (int r = 0; r < 8; ++r) {
        const int grow = m_blk + wrow + mb + r;
        out[(size_t)(b * L_SZ + grow) * D_SZ + h * E_SZ + wcol + n]      = c0[r];
        out[(size_t)(b * L_SZ + grow) * D_SZ + h * E_SZ + wcol + 16 + n] = c1[r];
    }
}

// ---------------- Host launch ----------------------------------------------
extern "C" void kernel_launch(void* const* d_in, const int* in_sizes, int n_in,
                              void* d_out, int out_size, void* d_ws, size_t ws_size,
                              hipStream_t stream) {
    const float* queries = (const float*)d_in[0];
    const float* keys    = (const float*)d_in[1];
    const float* values  = (const float*)d_in[2];
    const float* Wq = (const float*)d_in[3];
    const float* bq = (const float*)d_in[4];
    const float* Wk = (const float*)d_in[5];
    const float* bk = (const float*)d_in[6];
    const float* Wv = (const float*)d_in[7];
    const float* bv = (const float*)d_in[8];

    float* out_ptr  = (float*)d_out;                               // [B,L,D]
    float* attn_ptr = out_ptr + (size_t)B_SZ * L_SZ * D_SZ;        // [B,H,L,L]

    const size_t bld = (size_t)B_SZ * L_SZ * D_SZ;
    _Float16* qws = (_Float16*)d_ws;   // [B,L,H*E] f16, 8 MB
    _Float16* kws = qws + bld;         // [B,L,H*E] f16, 8 MB
    _Float16* vtw = kws + bld;         // [B*H, E, L] f16 (transposed), 8 MB
    _Float16* wtw = vtw + bld;         // 3 x [D][D] f16 transposed, 1.5 MB

    dim3 g0(D_SZ / 64, D_SZ / 64, 3);
    wt_kernel<<<g0, 256, 0, stream>>>(Wq, Wk, Wv, wtw);

    dim3 g1(B_SZ * L_SZ / 128, D_SZ / 64, 3);
    qkv_proj_kernel<<<g1, 256, 0, stream>>>(queries, keys, values, wtw,
                                            bq, bk, bv, qws, kws, vtw);

    dim3 g2(L_SZ / 64, L_SZ / 64, B_SZ * H_SZ);
    scores_kernel<<<g2, 256, 0, stream>>>(qws, kws, attn_ptr);

    softmax_kernel<<<dim3(B_SZ * H_SZ * L_SZ), 256, 0, stream>>>(attn_ptr);

    dim3 g4(L_SZ / 64, B_SZ * H_SZ);
    av_kernel<<<g4, 256, 0, stream>>>(attn_ptr, vtw, out_ptr);
}
